// PiGNNLayer_24300924961382
// MI455X (gfx1250) — matrix operand order repaired and checked by reading, not compiled, not run on hardware
//
#include <hip/hip_runtime.h>
#include <hip/hip_bf16.h>
#include <math.h>

// ---------------------------------------------------------------- constants
#define NN      20000        // nodes
#define KK      30           // neighbors per node
#define DD      128          // feature dim
#define HH      4            // heads
#define DH      32           // dim per head
#define EE      600000       // edges (NN*KK)
#define NT_EDGE 37500        // EE/16 edge M-tiles
#define NT_NODE 1250         // NN/16 node M-tiles
#define BB      16           // graphs

// f16 transposed-weight region offsets (in halves). Layout: Wt[n*din + k].
#define OFF_ATT_W1   0        // 128x384
#define OFF_ATT_W2   49152    // 128x128
#define OFF_ATT_W3   65536    // 16x128 (padded from 4x128)
#define OFF_NODE_W1  67584    // 128x256
#define OFF_NODE_W2  100352   // 128x128
#define OFF_NODE_W3  116736   // 32x128
#define OFF_EDGE_W1  120832   // 128x384
#define OFF_EDGE_W2  169984   // 128x128
#define OFF_EDGE_W3  186368   // 128x128
#define OFF_TOH_W    202752   // 128x128
#define OFF_GATE_W1  219136   // 128x128
#define OFF_GATE_W2  235520   // 128x128
#define OFF_GATE_W3  251904   // 128x128
#define W_TOTAL_HALF 268288

typedef __attribute__((ext_vector_type(16))) _Float16 v16h;
typedef __attribute__((ext_vector_type(4)))  _Float16 v4h;
typedef __attribute__((ext_vector_type(8)))  float    v8f;

// ---------------------------------------------------------------- helpers
__device__ __forceinline__ float reluf(float x) { return x > 0.f ? x : 0.f; }
__device__ __forceinline__ float geluf(float x) {
  return 0.5f * x * (1.f + erff(x * 0.70710678118654752f));
}

// A fragment: 16x32 f16 tile, row-major with stride lda (halves).
// ISA 7.12.2 (16-bit A 16x32): lane L -> row = L&15; lanes>=16 K-group +8;
// VGPR i (elems 2i,2i+1): K = (i&4?16:0) + 8*(L>>4) + 2*(i&3) + {0,1}.
__device__ __forceinline__ v16h load_a16x32(const _Float16* base, int lda) {
  const int lane = threadIdx.x & 31;
  const int row  = lane & 15;
  const int kg   = (lane >> 4) << 3;   // 0 or 8
  union { v16h v; unsigned int u[8]; } f;
  const _Float16* rp = base + (size_t)row * lda;
#pragma unroll
  for (int i = 0; i < 8; ++i) {
    const int k0 = ((i & 4) ? 16 : 0) + kg + ((i & 3) << 1);
    f.u[i] = *(const unsigned int*)(rp + k0);
  }
  return f.v;
}

// B fragment: 32x16 f16 tile of weight Wt (transposed: Wt[n*din + k]).
// lane L -> col N = ncol0 + (L&15); elems e<8 hold K = kc + 8*(L>>4) + e,
// e>=8 hold K = kc + 16 + 8*(L>>4) + (e-8) (two b128 loads, L2-resident weights).
__device__ __forceinline__ v16h load_b32x16(const _Float16* wt, int din,
                                            int kc, int ncol0) {
  const int lane = threadIdx.x & 31;
  const int n    = lane & 15;
  const int kg   = (lane >> 4) << 3;   // 0 or 8
  union { v16h v; uint4 q[2]; } f;
  const uint4* p = (const uint4*)(wt + (size_t)(ncol0 + n) * din + kc + kg);
  f.q[0] = p[0];
  f.q[1] = p[2];
  return f.v;
}

// D = A[16 x DIN] * Wt^T[DIN x NT*16], f32 accum, v_wmma_f32_16x16x32_f16.
// All NT B-fragments of one K-slab are loaded into distinct registers BEFORE
// the WMMA burst so the compiler can clause the b128 loads and stagger
// s_wait_loadcnt instead of draining to zero per WMMA (MLP: memory-level
// parallelism against L2-resident weights).
template <int DIN, int NT>
__device__ __forceinline__ void gemm_tile(const _Float16* A, int lda,
                                          const _Float16* Wt, v8f* acc) {
  const v8f vz = {};
#pragma unroll
  for (int nc = 0; nc < NT; ++nc) acc[nc] = vz;
#pragma unroll
  for (int kc = 0; kc < DIN / 32; ++kc) {
    v16h a = load_a16x32(A + kc * 32, lda);
    v16h b[NT];
#pragma unroll
    for (int nc = 0; nc < NT; ++nc)
      b[nc] = load_b32x16(Wt, DIN, kc * 32, nc * 16);
    if (kc + 1 < DIN / 32)
      __builtin_prefetch(Wt + kc * 32 + 32, 0, 1);   // global_prefetch_b8
#pragma unroll
    for (int nc = 0; nc < NT; ++nc)
      acc[nc] = __builtin_amdgcn_wmma_f32_16x16x32_f16(
          false, a, false, b[nc], (short)0, acc[nc], false, false);
  }
}

// C/D layout (f32 16x16): lane L col = L&15, rows r + 8*(L>>4) for VGPR r.
template <int NT, int ACT>  // ACT: 1=relu, 2=gelu
__device__ __forceinline__ void epilogue_lds(const v8f* acc, const float* bias,
                                             _Float16* out, int ldo) {
  const int lane  = threadIdx.x & 31;
  const int col0  = lane & 15;
  const int rbase = (lane >> 4) << 3;
#pragma unroll
  for (int nc = 0; nc < NT; ++nc) {
    const int col = nc * 16 + col0;
    const float b = bias[col];
#pragma unroll
    for (int r = 0; r < 8; ++r) {
      float x = acc[nc][r] + b;
      x = (ACT == 1) ? reluf(x) : geluf(x);
      out[(size_t)(rbase + r) * ldo + col] = (_Float16)x;
    }
  }
}

// ---------------------------------------------------------------- kernels
__global__ void zero_kernel(float* p, int n) {
  int i = blockIdx.x * blockDim.x + threadIdx.x;
  if (i < n) p[i] = 0.f;
}

// out[n*din + k] = (n < dout) ? w[k*dout + n] : 0   (f32 -> f16 transpose)
__global__ void wtrans_kernel(const float* __restrict__ w,
                              _Float16* __restrict__ out,
                              int din, int dout, int dout_pad) {
  int i = blockIdx.x * blockDim.x + threadIdx.x;
  if (i >= din * dout_pad) return;
  int n = i / din, k = i - n * din;
  out[(size_t)n * din + k] = (n < dout) ? (_Float16)w[(size_t)k * dout + n]
                                        : (_Float16)0.f;
}

__global__ void count_kernel(const long long* __restrict__ bidx, float* cnt) {
  int n = blockIdx.x * blockDim.x + threadIdx.x;
  if (n < NN) atomicAdd(&cnt[bidx[n]], 1.f);
}

// ---- fused per-edge MLPs: attention logits, node values, edge update -----
__global__ __launch_bounds__(64) void edge_mlps_kernel(
    const float* __restrict__ h, const float* __restrict__ e,
    const long long* __restrict__ srcI, const long long* __restrict__ dstI,
    const _Float16* __restrict__ W,
    const float* __restrict__ att_b1, const float* __restrict__ att_b2,
    const float* __restrict__ att_b3,
    const float* __restrict__ node_b1, const float* __restrict__ node_b2,
    const float* __restrict__ node_b3,
    const float* __restrict__ edge_b1, const float* __restrict__ edge_b2,
    const float* __restrict__ edge_b3,
    float* __restrict__ logits, _Float16* __restrict__ v16o,
    float* __restrict__ e_out, float* __restrict__ bn_sum,
    float* __restrict__ bn_sq) {
  __shared__ __align__(16) _Float16 sCat3[2][16 * 384];
  __shared__ __align__(16) _Float16 sP[2][16 * 128];
  __shared__ __align__(16) _Float16 sQ[2][16 * 128];

  const int wv   = threadIdx.x >> 5;
  const int lane = threadIdx.x & 31;
  const long long t = (long long)blockIdx.x * 2 + wv;   // 37500 tiles exactly
  _Float16* cat3 = sCat3[wv];
  _Float16* a1   = sP[wv];
  _Float16* a2   = sQ[wv];
  const long long ebase = t * 16;

  // ---- build cat3 = [hi | e | hj] as f16 (wave-private rows, no barrier)
  const int c4 = lane << 2;   // 32 lanes x 4 cols = 128
  for (int r = 0; r < 16; ++r) {
    const long long g  = ebase + r;
    const long long si = srcI[g];
    const long long di = dstI[g];
    const float4 hv = *(const float4*)(h + si * 128 + c4);
    const float4 ev = *(const float4*)(e + g * 128 + c4);
    const float4 jv = *(const float4*)(h + di * 128 + c4);
    _Float16* row = cat3 + r * 384;
    *(v4h*)(row + c4)       = v4h{(_Float16)hv.x, (_Float16)hv.y,
                                  (_Float16)hv.z, (_Float16)hv.w};
    *(v4h*)(row + 128 + c4) = v4h{(_Float16)ev.x, (_Float16)ev.y,
                                  (_Float16)ev.z, (_Float16)ev.w};
    *(v4h*)(row + 256 + c4) = v4h{(_Float16)jv.x, (_Float16)jv.y,
                                  (_Float16)jv.z, (_Float16)jv.w};
  }

  const int col0  = lane & 15;
  const int rbase = (lane >> 4) << 3;
  v8f acc[8];

  // ---------------- attention MLP: relu, relu, linear/sqrt(DH) ----------
  gemm_tile<384, 8>(cat3, 384, W + OFF_ATT_W1, acc);
  epilogue_lds<8, 1>(acc, att_b1, a1, 128);
  gemm_tile<128, 8>(a1, 128, W + OFF_ATT_W2, acc);
  epilogue_lds<8, 1>(acc, att_b2, a2, 128);
  {
    v8f l[1];
    gemm_tile<128, 1>(a2, 128, W + OFF_ATT_W3, l);   // padded 4->16 cols
    const float b = (col0 < 4) ? att_b3[col0] : 0.f;
#pragma unroll
    for (int r = 0; r < 8; ++r) {
      float x = (l[0][r] + b) * 0.17677669529663687f;  // 1/sqrt(32)
      if (col0 < 4) logits[(ebase + rbase + r) * 4 + col0] = x;
    }
  }

  // ---------------- node value MLP: gelu, gelu, linear (cat2 view) -------
  gemm_tile<256, 8>(cat3 + 128, 384, W + OFF_NODE_W1, acc);
  epilogue_lds<8, 2>(acc, node_b1, a1, 128);
  gemm_tile<128, 8>(a1, 128, W + OFF_NODE_W2, acc);
  epilogue_lds<8, 2>(acc, node_b2, a2, 128);
  {
    v8f vv[2];
    gemm_tile<128, 2>(a2, 128, W + OFF_NODE_W3, vv);
#pragma unroll
    for (int nc = 0; nc < 2; ++nc) {
      const int col = nc * 16 + col0;
      const float b = node_b3[col];
#pragma unroll
      for (int r = 0; r < 8; ++r)
        v16o[(ebase + rbase + r) * 32 + col] = (_Float16)(vv[nc][r] + b);
    }
  }

  // ---------------- edge MLP + residual + BN partial sums ---------------
  gemm_tile<384, 8>(cat3, 384, W + OFF_EDGE_W1, acc);
  epilogue_lds<8, 2>(acc, edge_b1, a1, 128);
  gemm_tile<128, 8>(a1, 128, W + OFF_EDGE_W2, acc);
  epilogue_lds<8, 2>(acc, edge_b2, a2, 128);
  gemm_tile<128, 8>(a2, 128, W + OFF_EDGE_W3, acc);
  {
#pragma unroll
    for (int nc = 0; nc < 8; ++nc) {
      const int col = nc * 16 + col0;
      const float b = edge_b3[col];
      float s = 0.f, q = 0.f;
#pragma unroll
      for (int r = 0; r < 8; ++r) {
        const long long g = ebase + rbase + r;
        float x = acc[nc][r] + b + e[g * 128 + col];
        e_out[g * 128 + col] = x;    // pre-BN, normalized in a later pass
        s += x;
        q += x * x;
      }
      s += __shfl_xor(s, 16, 32);    // lanes L, L+16 share a column
      q += __shfl_xor(q, 16, 32);
      if (lane < 16) {
        atomicAdd(&bn_sum[col], s);
        atomicAdd(&bn_sq[col], q);
      }
    }
  }
}

// ---- per-node softmax over K + attention-weighted value aggregation -----
__global__ __launch_bounds__(256) void aggregate_kernel(
    const float* __restrict__ logits, const _Float16* __restrict__ v16i,
    _Float16* __restrict__ hpre) {
  __shared__ float satt[8][128];
  const int wv   = threadIdx.x >> 5;
  const int lane = threadIdx.x & 31;
  const int n    = blockIdx.x * 8 + wv;            // 2500*8 = 20000 exactly
  const long long eb = (long long)n * KK;

  for (int hh = 0; hh < HH; ++hh) {
    float x = (lane < KK) ? logits[(eb + lane) * 4 + hh] : -3.4e38f;
    float m = x;
    for (int o = 16; o; o >>= 1) m = fmaxf(m, __shfl_xor(m, o, 32));
    float ex = (lane < KK) ? __expf(x - m) : 0.f;
    float s = ex;
    for (int o = 16; o; o >>= 1) s += __shfl_xor(s, o, 32);
    satt[wv][hh * 32 + lane] = ex / s;
  }
  __syncthreads();

  const int hh = lane >> 3;        // head of this lane's 4 features
  const int f0 = lane << 2;
  const int d0 = f0 & 31;
  float a0 = 0.f, a1 = 0.f, a2 = 0.f, a3 = 0.f;
  for (int k = 0; k < KK; ++k) {
    const float w = satt[wv][hh * 32 + k];
    const _Float16* vp = v16i + (eb + k) * 32;
    a0 += w * (float)vp[d0 + 0];
    a1 += w * (float)vp[d0 + 1];
    a2 += w * (float)vp[d0 + 2];
    a3 += w * (float)vp[d0 + 3];
  }
  _Float16* op = hpre + (long long)n * 128 + f0;
  op[0] = (_Float16)a0; op[1] = (_Float16)a1;
  op[2] = (_Float16)a2; op[3] = (_Float16)a3;
}

// ---- _h = hpre @ toh_w (WMMA) + per-graph segment sums ------------------
__global__ __launch_bounds__(64) void toh_kernel(
    const _Float16* __restrict__ hpre, const _Float16* __restrict__ Wt,
    const long long* __restrict__ bidx, float* __restrict__ htoh,
    float* __restrict__ csum) {
  const int wv   = threadIdx.x >> 5;
  const int lane = threadIdx.x & 31;
  const int t    = blockIdx.x * 2 + wv;            // 625*2 = 1250 exactly
  v8f acc[8];
  gemm_tile<128, 8>(hpre + (size_t)t * 16 * 128, 128, Wt, acc);
  const int col0  = lane & 15;
  const int rbase = (lane >> 4) << 3;
#pragma unroll
  for (int nc = 0; nc < 8; ++nc) {
    const int col = nc * 16 + col0;
#pragma unroll
    for (int r = 0; r < 8; ++r) {
      const long long n = (long long)t * 16 + rbase + r;
      const float x = acc[nc][r];
      htoh[n * 128 + col] = x;
      atomicAdd(&csum[bidx[n] * 128 + col], x);
    }
  }
}

// ---- gate MLP on [16,128] graph means: one block, one WMMA N-tile/wave --
__global__ __launch_bounds__(256) void gate_kernel(
    const float* __restrict__ csum, const float* __restrict__ cnt,
    const _Float16* __restrict__ W, const float* __restrict__ gb1,
    const float* __restrict__ gb2, const float* __restrict__ gb3,
    float* __restrict__ gate) {
  __shared__ __align__(16) _Float16 sc[16 * 128];
  __shared__ __align__(16) _Float16 s1[16 * 128];
  __shared__ __align__(16) _Float16 s2[16 * 128];
  const int wv   = threadIdx.x >> 5;
  const int lane = threadIdx.x & 31;
  for (int i = threadIdx.x; i < 2048; i += 256)
    sc[i] = (_Float16)(csum[i] / cnt[i >> 7]);
  __syncthreads();

  const int col0  = lane & 15;
  const int rbase = (lane >> 4) << 3;
  const int col   = wv * 16 + col0;
  v8f acc[1];

  gemm_tile<128, 1>(sc, 128, W + OFF_GATE_W1 + wv * 2048, acc);
#pragma unroll
  for (int r = 0; r < 8; ++r)
    s1[(rbase + r) * 128 + col] = (_Float16)reluf(acc[0][r] + gb1[col]);
  __syncthreads();

  gemm_tile<128, 1>(s1, 128, W + OFF_GATE_W2 + wv * 2048, acc);
#pragma unroll
  for (int r = 0; r < 8; ++r)
    s2[(rbase + r) * 128 + col] = (_Float16)reluf(acc[0][r] + gb2[col]);
  __syncthreads();

  gemm_tile<128, 1>(s2, 128, W + OFF_GATE_W3 + wv * 2048, acc);
#pragma unroll
  for (int r = 0; r < 8; ++r) {
    const float x = acc[0][r] + gb3[col];
    gate[(rbase + r) * 128 + col] = 1.f / (1.f + expf(-x));
  }
}

__global__ void hout_kernel(const float* __restrict__ htoh,
                            const float* __restrict__ gate,
                            const long long* __restrict__ bidx,
                            float* __restrict__ out) {
  const long long i = (long long)blockIdx.x * blockDim.x + threadIdx.x;
  if (i >= (long long)NN * 128) return;
  const long long n = i >> 7;
  out[i] = htoh[i] * gate[bidx[n] * 128 + (i & 127)];
}

__global__ void bn_stats_kernel(const float* __restrict__ bn_sum,
                                const float* __restrict__ bn_sq,
                                float* __restrict__ mu,
                                float* __restrict__ rstd) {
  const int c = threadIdx.x;
  if (c < 128) {
    const float inv = 1.f / (float)EE;
    const float m = bn_sum[c] * inv;
    const float v = bn_sq[c] * inv - m * m;
    mu[c] = m;
    rstd[c] = rsqrtf(v + 1e-5f);
  }
}

__global__ void bn_apply_kernel(float* __restrict__ e_out,
                                const float* __restrict__ mu,
                                const float* __restrict__ rstd,
                                const float* __restrict__ g,
                                const float* __restrict__ b) {
  const long long i = (long long)blockIdx.x * blockDim.x + threadIdx.x;
  if (i >= (long long)EE * 128) return;
  const int c = i & 127;
  e_out[i] = (e_out[i] - mu[c]) * rstd[c] * g[c] + b[c];
}

// ---------------------------------------------------------------- launch
extern "C" void kernel_launch(void* const* d_in, const int* in_sizes, int n_in,
                              void* d_out, int out_size, void* d_ws,
                              size_t ws_size, hipStream_t stream) {
  (void)in_sizes; (void)n_in; (void)out_size; (void)ws_size;
  const float* h       = (const float*)d_in[0];
  const float* e       = (const float*)d_in[1];
  const float* att_w1  = (const float*)d_in[2];
  const float* att_b1  = (const float*)d_in[3];
  const float* att_w2  = (const float*)d_in[4];
  const float* att_b2  = (const float*)d_in[5];
  const float* att_w3  = (const float*)d_in[6];
  const float* att_b3  = (const float*)d_in[7];
  const float* node_w1 = (const float*)d_in[8];
  const float* node_b1 = (const float*)d_in[9];
  const float* node_w2 = (const float*)d_in[10];
  const float* node_b2 = (const float*)d_in[11];
  const float* node_w3 = (const float*)d_in[12];
  const float* node_b3 = (const float*)d_in[13];
  const float* toh_w   = (const float*)d_in[14];
  const float* edge_w1 = (const float*)d_in[15];
  const float* edge_b1 = (const float*)d_in[16];
  const float* edge_w2 = (const float*)d_in[17];
  const float* edge_b2 = (const float*)d_in[18];
  const float* edge_w3 = (const float*)d_in[19];
  const float* edge_b3 = (const float*)d_in[20];
  const float* bn_g    = (const float*)d_in[21];
  const float* bn_b    = (const float*)d_in[22];
  const float* gate_w1 = (const float*)d_in[23];
  const float* gate_b1 = (const float*)d_in[24];
  const float* gate_w2 = (const float*)d_in[25];
  const float* gate_b2 = (const float*)d_in[26];
  const float* gate_w3 = (const float*)d_in[27];
  const float* gate_b3 = (const float*)d_in[28];
  const long long* eidx = (const long long*)d_in[29];
  const long long* bidx = (const long long*)d_in[30];
  const long long* srcI = eidx;
  const long long* dstI = eidx + (long long)EE;

  float* out_h = (float*)d_out;
  float* out_e = out_h + (size_t)NN * 128;   // pre-BN written here, BN in place

  // ---- workspace carve-up (256B aligned slabs) ----
  char* ws = (char*)d_ws;
  size_t off = 0;
  auto take = [&](size_t bytes) {
    size_t r = off;
    off += (bytes + 255) & ~(size_t)255;
    return r;
  };
  _Float16* W16   = (_Float16*)(ws + take(W_TOTAL_HALF * 2));
  float*    logits= (float*)   (ws + take((size_t)EE * 4 * 4));
  _Float16* v16   = (_Float16*)(ws + take((size_t)EE * 32 * 2));
  _Float16* hpre  = (_Float16*)(ws + take((size_t)NN * 128 * 2));
  float*    htoh  = (float*)   (ws + take((size_t)NN * 128 * 4));
  float*    zblk  = (float*)   (ws + take(2320 * 4));
  float*    bn_sum = zblk;            // 128
  float*    bn_sq  = zblk + 128;      // 128
  float*    csum   = zblk + 256;      // 16*128
  float*    cnt    = zblk + 2304;     // 16
  float*    mu    = (float*)(ws + take(128 * 4));
  float*    rstd  = (float*)(ws + take(128 * 4));
  float*    gate  = (float*)(ws + take(16 * 128 * 4));

  // ---- 0) zero accumulators ----
  zero_kernel<<<(2320 + 255) / 256, 256, 0, stream>>>(zblk, 2320);

  // ---- 1) transpose+convert weights to f16 ----
  auto wt = [&](const float* w, int doff, int din, int dout, int dpad) {
    int n = din * dpad;
    wtrans_kernel<<<(n + 255) / 256, 256, 0, stream>>>(w, W16 + doff, din,
                                                       dout, dpad);
  };
  wt(att_w1,  OFF_ATT_W1,  384, 128, 128);
  wt(att_w2,  OFF_ATT_W2,  128, 128, 128);
  wt(att_w3,  OFF_ATT_W3,  128,   4,  16);
  wt(node_w1, OFF_NODE_W1, 256, 128, 128);
  wt(node_w2, OFF_NODE_W2, 128, 128, 128);
  wt(node_w3, OFF_NODE_W3, 128,  32,  32);
  wt(edge_w1, OFF_EDGE_W1, 384, 128, 128);
  wt(edge_w2, OFF_EDGE_W2, 128, 128, 128);
  wt(edge_w3, OFF_EDGE_W3, 128, 128, 128);
  wt(toh_w,   OFF_TOH_W,   128, 128, 128);
  wt(gate_w1, OFF_GATE_W1, 128, 128, 128);
  wt(gate_w2, OFF_GATE_W2, 128, 128, 128);
  wt(gate_w3, OFF_GATE_W3, 128, 128, 128);

  // ---- 2) per-graph node counts ----
  count_kernel<<<(NN + 255) / 256, 256, 0, stream>>>(bidx, cnt);

  // ---- 3) fused per-edge MLPs (the WMMA heavy hitter) ----
  edge_mlps_kernel<<<NT_EDGE / 2, 64, 0, stream>>>(
      h, e, srcI, dstI, W16, att_b1, att_b2, att_b3, node_b1, node_b2,
      node_b3, edge_b1, edge_b2, edge_b3, logits, v16, out_e, bn_sum, bn_sq);

  // ---- 4) softmax + aggregation -> hpre (f16) ----
  aggregate_kernel<<<NN / 8, 256, 0, stream>>>(logits, v16, hpre);

  // ---- 5) _h = hpre @ toh_w + segment sums ----
  toh_kernel<<<NT_NODE / 2, 64, 0, stream>>>(hpre, W16 + OFF_TOH_W, bidx,
                                             htoh, csum);

  // ---- 6) gate MLP on graph means ----
  gate_kernel<<<1, 256, 0, stream>>>(csum, cnt, W16, gate_b1, gate_b2,
                                     gate_b3, gate);

  // ---- 7) h_out = _h * g[batch] ----
  hout_kernel<<<(NN * 128) / 256, 256, 0, stream>>>(htoh, gate, bidx, out_h);

  // ---- 8) batchnorm finalize + apply (in place on out_e) ----
  bn_stats_kernel<<<1, 128, 0, stream>>>(bn_sum, bn_sq, mu, rstd);
  bn_apply_kernel<<<((size_t)EE * 128) / 256, 256, 0, stream>>>(out_e, mu,
                                                                rstd, bn_g,
                                                                bn_b);
}
